// LinearApproximation_55224689492189
// MI455X (gfx1250) — matrix operand order repaired and checked by reading, not compiled, not run on hardware
//
#include <hip/hip_runtime.h>

// ---------------------------------------------------------------------------
// Y = X @ W^T + b  via bf16x3 split-precision WMMA on gfx1250 (wave32).
//   X: [B,512] f32   W: [512,512] f32 (row-major, [out][in])   b: [512] f32
// Block tile: 128(M) x 128(N), K streamed in chunks of 32.
// 256 threads = 8 waves arranged 4(M) x 2(N); each wave computes 32x64
// (2 mt x 4 nt accumulators) -> 24 WMMAs per wave per K-step.
// Split: hi = trunc16(x), lo = trunc16(x - up(hi)); packed with v_perm_b32
// (3 VALU ops per float). A*B ~= Ah*Bh + Ah*Bl + Al*Bh, f32 accumulate.
// ---------------------------------------------------------------------------

typedef __attribute__((ext_vector_type(16))) __bf16 v16bf;
typedef __attribute__((ext_vector_type(8)))  __bf16 v8bf;
typedef __attribute__((ext_vector_type(8)))  float  v8f;
typedef __attribute__((ext_vector_type(4)))  float  v4f;
typedef __attribute__((ext_vector_type(2)))  unsigned v2u;

#define BM    128
#define BN    128
#define BK    32
#define LDSW  40          // padded LDS row stride (bf16 elems): 80B, 16B-aligned
#define KTOT  512
#define NTOT  512

// Pack the high 16 bits of two f32 (a -> low half, b -> high half) in one op.
__device__ __forceinline__ unsigned pack_hi16(float a, float b) {
    // dst byte0=a.b2, byte1=a.b3, byte2=b.b2, byte3=b.b3  (S1 = a, S0 = b)
    return __builtin_amdgcn_perm(__float_as_uint(b), __float_as_uint(a), 0x07060302u);
}
__device__ __forceinline__ float trunc_resid(float x) {
    return x - __uint_as_float(__float_as_uint(x) & 0xFFFF0000u);
}
__device__ __forceinline__ void split4(const v4f x, v2u& hi, v2u& lo) {
    hi = (v2u){ pack_hi16(x[0], x[1]), pack_hi16(x[2], x[3]) };
    float r0 = trunc_resid(x[0]), r1 = trunc_resid(x[1]);
    float r2 = trunc_resid(x[2]), r3 = trunc_resid(x[3]);
    lo = (v2u){ pack_hi16(r0, r1), pack_hi16(r2, r3) };
}

union FragU { v16bf v; v8bf h[2]; };

// A fragment 16x32 bf16: lane<16 holds K[kh*8 .. +8) and K[kh*8+16 .. +8) of row M=lane%16
__device__ __forceinline__ v16bf load_fragA(const unsigned short* plane, int row, int kh) {
    const unsigned short* p = plane + row * LDSW + kh * 8;
    FragU f;
    f.h[0] = *(const v8bf*)(p);
    f.h[1] = *(const v8bf*)(p + 16);
    return f.v;
}
// B fragment 32x16 bf16: lane holds K[kh*8 .. +8) and K[kh*8+16 .. +8) of column N=lane%16
__device__ __forceinline__ v16bf load_fragB(const unsigned short* plane, int n, int kh) {
    const unsigned short* p = plane + n * LDSW + kh * 8;
    FragU f;
    f.h[0] = *(const v8bf*)(p);
    f.h[1] = *(const v8bf*)(p + 16);
    return f.v;
}

__global__ __launch_bounds__(256)
void linear_bf16x3_wmma(const float* __restrict__ X,
                        const float* __restrict__ W,
                        const float* __restrict__ bias,
                        float* __restrict__ Y)
{
    __shared__ __align__(16) unsigned short sAh[BM * LDSW];
    __shared__ __align__(16) unsigned short sAl[BM * LDSW];
    __shared__ __align__(16) unsigned short sBh[BN * LDSW];
    __shared__ __align__(16) unsigned short sBl[BN * LDSW];

    const int tid   = threadIdx.x;
    const int lane  = tid & 31;
    const int wid   = tid >> 5;
    const int wm    = wid & 3;          // 4 waves along M (32 rows each)
    const int wn    = wid >> 2;         // 2 waves along N (64 cols each)
    const int m0    = blockIdx.y * BM;
    const int n0    = blockIdx.x * BN;
    const int l16   = lane & 15;
    const int lhalf = lane >> 4;

    // ---- loop-invariant staging addresses (loads use immediate offsets) ----
    const int rowS = tid >> 3;          // 0..31 (each j adds 32 rows)
    const int kq   = (tid & 7) << 2;    // float4 column within the K-chunk
    const float* xp = X + (size_t)(m0 + rowS) * KTOT + kq;
    const float* wp = W + (size_t)(n0 + rowS) * KTOT + kq;
    unsigned short* aH = sAh + rowS * LDSW + kq;
    unsigned short* aL = sAl + rowS * LDSW + kq;
    unsigned short* bH = sBh + rowS * LDSW + kq;
    unsigned short* bL = sBl + rowS * LDSW + kq;

    v8f acc[2][4];
    #pragma unroll
    for (int mt = 0; mt < 2; ++mt)
        #pragma unroll
        for (int nt = 0; nt < 4; ++nt)
            acc[mt][nt] = (v8f){0.f,0.f,0.f,0.f,0.f,0.f,0.f,0.f};

    for (int k0 = 0; k0 < KTOT; k0 += BK) {
        // ---- stage X tile (128x32 f32) into hi/lo bf16 planes
        #pragma unroll
        for (int j = 0; j < 4; ++j) {
            v4f xv = *(const v4f*)(xp + j * 32 * KTOT);   // imm offset 65536*j B
            v2u hv, lv;
            split4(xv, hv, lv);
            *(v2u*)(aH + j * 32 * LDSW) = hv;             // imm offset 2560*j B
            *(v2u*)(aL + j * 32 * LDSW) = lv;
        }
        // ---- stage W tile (128x32 f32): column n of B == row n of W
        #pragma unroll
        for (int j = 0; j < 4; ++j) {
            v4f wv = *(const v4f*)(wp + j * 32 * KTOT);
            v2u hv, lv;
            split4(wv, hv, lv);
            *(v2u*)(bH + j * 32 * LDSW) = hv;
            *(v2u*)(bL + j * 32 * LDSW) = lv;
        }
        xp += BK;
        wp += BK;
        // ---- prefetch next K-chunk (global_prefetch_b8), exact lines needed
        if (k0 + BK < KTOT) {
            __builtin_prefetch(xp, 0, 0);
            __builtin_prefetch(wp, 0, 0);
        }
        __syncthreads();

        // ---- fragments for this wave's 32x64 patch
        v16bf ah[2], al[2], bh2[4], bl2[4];
        #pragma unroll
        for (int t = 0; t < 2; ++t) {
            ah[t] = load_fragA(sAh, wm * 32 + t * 16 + l16, lhalf);
            al[t] = load_fragA(sAl, wm * 32 + t * 16 + l16, lhalf);
        }
        #pragma unroll
        for (int t = 0; t < 4; ++t) {
            bh2[t] = load_fragB(sBh, wn * 64 + t * 16 + l16, lhalf);
            bl2[t] = load_fragB(sBl, wn * 64 + t * 16 + l16, lhalf);
        }

        // ---- bf16x3: Ah*Bh + Ah*Bl + Al*Bh, fp32 accumulate (24 WMMAs)
        #pragma unroll
        for (int mt = 0; mt < 2; ++mt) {
            #pragma unroll
            for (int nt = 0; nt < 4; ++nt) {
                acc[mt][nt] = __builtin_amdgcn_wmma_f32_16x16x32_bf16(
                    false, ah[mt], false, bh2[nt], (short)0, acc[mt][nt], false, false);
                acc[mt][nt] = __builtin_amdgcn_wmma_f32_16x16x32_bf16(
                    false, ah[mt], false, bl2[nt], (short)0, acc[mt][nt], false, false);
                acc[mt][nt] = __builtin_amdgcn_wmma_f32_16x16x32_bf16(
                    false, al[mt], false, bh2[nt], (short)0, acc[mt][nt], false, false);
            }
        }
        __syncthreads();
    }

    // ---- epilogue: C/D layout VGPR r <-> M = 8*(lane/16)+r, N = lane%16
    #pragma unroll
    for (int nt = 0; nt < 4; ++nt) {
        int col = n0 + wn * 64 + nt * 16 + l16;
        float bv = bias[col];
        #pragma unroll
        for (int mt = 0; mt < 2; ++mt) {
            int rbase = m0 + wm * 32 + mt * 16 + lhalf * 8;
            #pragma unroll
            for (int r = 0; r < 8; ++r) {
                Y[(size_t)(rbase + r) * NTOT + col] = acc[mt][nt][r] + bv;
            }
        }
    }
}

extern "C" void kernel_launch(void* const* d_in, const int* in_sizes, int n_in,
                              void* d_out, int out_size, void* d_ws, size_t ws_size,
                              hipStream_t stream) {
    const float* X    = (const float*)d_in[0];
    const float* W    = (const float*)d_in[1];
    const float* bias = (const float*)d_in[2];
    float* Y          = (float*)d_out;

    const int rows = in_sizes[0] / KTOT;          // 65536
    dim3 grid(NTOT / BN, rows / BM);              // (4, 512)
    linear_bf16x3_wmma<<<grid, 256, 0, stream>>>(X, W, bias, Y);
}
